// SEALModel_10007273800344
// MI455X (gfx1250) — matrix operand order repaired and checked by reading, not compiled, not run on hardware
//
#include <hip/hip_runtime.h>
#include <hip/hip_bf16.h>
#include <math.h>
#include <stdint.h>

// ---------------------------------------------------------------------------
// CDNA5 WMMA types
// ---------------------------------------------------------------------------
typedef __attribute__((ext_vector_type(16))) _Float16 v16h;
typedef __attribute__((ext_vector_type(8)))  float    v8f;

#define LEAK 0.01f
#define BN_EPS 1e-5f

static inline int ceil_div(int a, int b) { return (a + b - 1) / b; }

// ---------------------------------------------------------------------------
// WMMA GEMM: Out[N x Hout] = X[N x K] @ W[K x Hout] (+ Cin) (+ bias)
// fp32 in memory, f16 A/B operands, fp32 accumulate (v_wmma_f32_16x16x32_f16).
//
// Grid: (ceil(rowTiles/8), colTiles). One block = 8 waves; every wave computes
// one 16x16 output tile in the same 16-wide column strip, so the whole block
// shares one W panel (K x 16 fp32, <= 8 KB). The panel is staged into LDS with
// CDNA5 async global->LDS copies (ASYNCcnt), then B operands come from LDS.
// N multiple of 16, K multiple of 32, Hout multiple of 16 (true for all calls).
// ---------------------------------------------------------------------------
__global__ void wmma_gemm_f16(const float* __restrict__ X,
                              const float* __restrict__ W,
                              const float* __restrict__ Cin,
                              const float* __restrict__ bias,
                              float* __restrict__ Out,
                              int N, int K, int Hout) {
    __shared__ float sW[128 * 16];          // max K = 128 -> 8 KB panel
    const int lane = threadIdx.x & 31;
    const int wave = threadIdx.x >> 5;
    const int rowTiles = N >> 4;
    const int colBase = blockIdx.y << 4;

    // ---- stage W[:, colBase:colBase+16] into LDS via async copy ----
    {
        const unsigned ldsbase = (unsigned)(uintptr_t)(void*)sW; // LDS offset = addr[31:0]
        const int nvec = K << 2;            // number of float4 chunks (K*16/4)
        for (int i = threadIdx.x; i < nvec; i += blockDim.x) {
            int k  = i >> 2;                // K row
            int c4 = (i & 3) << 2;          // starting column within panel
            unsigned gvoff   = (unsigned)(((size_t)k * Hout + colBase + c4) * sizeof(float));
            unsigned lds_off = ldsbase + (unsigned)i * 16u;
            // GLOBAL_LOAD_ASYNC_TO_LDS_B128: copies 16 B from W + gvoff into LDS at lds_off
            asm volatile("global_load_async_to_lds_b128 %0, %1, %2"
                         :
                         : "v"(lds_off), "v"(gvoff), "s"(W)
                         : "memory");
        }
        asm volatile("s_wait_asynccnt 0" ::: "memory");
        __syncthreads();
    }

    const int rowTile = blockIdx.x * (blockDim.x >> 5) + wave;
    if (rowTile >= rowTiles) return;
    const int rowBase = rowTile << 4;
    const int half = lane >> 4;             // which 16-lane half of the wave
    const int m    = lane & 15;

    // C/D layout: VGPR j -> row rowBase + j + 8*half, col colBase + m
    v8f acc = {};
    if (Cin) {
#pragma unroll
        for (int j = 0; j < 8; ++j)
            acc[j] = Cin[(size_t)(rowBase + j + 8 * half) * Hout + colBase + m];
    }

    for (int kb = 0; kb < K; kb += 32) {
        v16h a, b;
        // A (16x32 f16): lane half 0: v0..3 -> K 0..7, v4..7 -> K 16..23
        //                lane half 1: v0..3 -> K 8..15, v4..7 -> K 24..31
        const float* arow = X + (size_t)(rowBase + m) * K + kb;
#pragma unroll
        for (int v = 0; v < 8; ++v) {
            int k0 = ((v >> 2) << 4) + (half << 3) + ((v & 3) << 1);
            float2 f2 = *(const float2*)(arow + k0);   // contiguous K pair
            a[2 * v]     = (_Float16)f2.x;
            a[2 * v + 1] = (_Float16)f2.y;
        }
        // B (32x16 f16) from LDS: lanes 0-15 hold K 0..7,16..23 of column m,
        // lanes 16-31 hold K 8..15,24..31
        const int krow = kb + (half << 3);
#pragma unroll
        for (int i = 0; i < 16; ++i)
            b[i] = (_Float16)sW[(krow + (i & 7) + ((i >> 3) << 4)) * 16 + m];

        acc = __builtin_amdgcn_wmma_f32_16x16x32_f16(
            /*neg_a=*/false, a, /*neg_b=*/false, b,
            /*c_mod=*/(short)0, acc, /*reuse_a=*/false, /*reuse_b=*/false);
    }

    const float bv = bias ? bias[colBase + m] : 0.0f;
#pragma unroll
    for (int j = 0; j < 8; ++j)
        Out[(size_t)(rowBase + j + 8 * half) * Hout + colBase + m] = acc[j] + bv;
}

// ---------------------------------------------------------------------------
// Degree / normalization precompute
// ---------------------------------------------------------------------------
__global__ void init_deg(float* deg, int N) {
    int i = blockIdx.x * blockDim.x + threadIdx.x;
    if (i < N) deg[i] = 1.0f;   // self-loop contribution
}
__global__ void edge_deg(const int* __restrict__ dst, float* deg, int E) {
    int e = blockIdx.x * blockDim.x + threadIdx.x;
    if (e < E) atomicAdd(&deg[dst[e]], 1.0f);
}
// in-place: deg -> dinv = rsqrt(deg);  invcnt = 1/max(deg-1,1) for SAGE mean
__global__ void finalize_deg(float* dinv, float* invcnt, int N) {
    int i = blockIdx.x * blockDim.x + threadIdx.x;
    if (i >= N) return;
    float d = dinv[i];               // >= 1 due to self loop
    dinv[i]   = rsqrtf(d);
    invcnt[i] = 1.0f / fmaxf(d - 1.0f, 1.0f);
}

// ---------------------------------------------------------------------------
// Generic elementwise helpers
// ---------------------------------------------------------------------------
__global__ void zero_f(float* p, int n) {
    int i = blockIdx.x * blockDim.x + threadIdx.x;
    if (i < n) p[i] = 0.0f;
}

// Edge scatter: Out[dst] += norm * H[src] ; norm = dinv[s]*dinv[d] (GCN) or 1 (SAGE sum)
__global__ void edge_scatter(const int* __restrict__ src, const int* __restrict__ dst,
                             const float* __restrict__ dinv,
                             const float* __restrict__ Hm, float* __restrict__ Out,
                             int E, int F) {
    int t = blockIdx.x * blockDim.x + threadIdx.x;
    int perEdge = F >> 2;
    int e = t / perEdge;
    if (e >= E) return;
    int j = (t - e * perEdge) << 2;
    int s = src[e], d = dst[e];
    float nrm = dinv ? dinv[s] * dinv[d] : 1.0f;
    const float4 v = *(const float4*)(Hm + (size_t)s * F + j);
    float* o = Out + (size_t)d * F + j;
    atomicAdd(o + 0, nrm * v.x);
    atomicAdd(o + 1, nrm * v.y);
    atomicAdd(o + 2, nrm * v.z);
    atomicAdd(o + 3, nrm * v.w);
}

// GCN self-loop + bias: Out[i,c] += dinv[i]^2 * H[i,c] + b[c]
__global__ void gcn_self_bias(const float* __restrict__ dinv, const float* __restrict__ Hm,
                              const float* __restrict__ bias, float* __restrict__ Out,
                              int N, int F) {
    int i = blockIdx.x * blockDim.x + threadIdx.x;
    if (i >= N * F) return;
    int r = i / F, c = i - r * F;
    float di = dinv[r];
    Out[i] += di * di * Hm[i] + bias[c];
}

// SAGE mean: agg[i,:] *= invcnt[i]
__global__ void sage_scale(float* __restrict__ agg, const float* __restrict__ invcnt,
                           int N, int F) {
    int i = blockIdx.x * blockDim.x + threadIdx.x;
    if (i >= N * F) return;
    agg[i] *= invcnt[i / F];
}

// ---------------------------------------------------------------------------
// BatchNorm (training forward, biased variance) + LeakyReLU
// ---------------------------------------------------------------------------
__global__ void bn_stats(const float* __restrict__ X, float* sums, float* sumsq,
                         int N, int F) {
    int c = threadIdx.x % F;                       // F in {32,64,128} divides 256
    int rowsPerBlock = blockDim.x / F;
    int r0 = blockIdx.x * rowsPerBlock + threadIdx.x / F;
    int stride = gridDim.x * rowsPerBlock;
    float s = 0.0f, q = 0.0f;
    for (int r = r0; r < N; r += stride) {
        float v = X[(size_t)r * F + c];
        s += v;
        q += v * v;
    }
    atomicAdd(&sums[c], s);
    atomicAdd(&sumsq[c], q);
}

__global__ void bn_apply_lrelu(float* __restrict__ X, const float* __restrict__ sums,
                               const float* __restrict__ sumsq,
                               const float* __restrict__ g, const float* __restrict__ b,
                               int N, int F) {
    int i = blockIdx.x * blockDim.x + threadIdx.x;
    if (i >= N * F) return;
    int c = i % F;
    float invN = 1.0f / (float)N;
    float m = sums[c] * invN;
    float v = sumsq[c] * invN - m * m;
    float y = (X[i] - m) * rsqrtf(v + BN_EPS) * g[c] + b[c];
    X[i] = (y >= 0.0f) ? y : LEAK * y;
}

// ---------------------------------------------------------------------------
// Pool + link prediction
// ---------------------------------------------------------------------------
__global__ void pool_add(const float* __restrict__ h, const int* __restrict__ batch,
                         float* __restrict__ g, int N, int F) {
    int i = blockIdx.x * blockDim.x + threadIdx.x;
    if (i >= N * F) return;
    int r = i / F, c = i - r * F;
    atomicAdd(&g[(size_t)batch[r] * F + c], h[i]);
}

__global__ void link_pred(const float* __restrict__ g, const int* __restrict__ li,
                          float* __restrict__ out, int L, int F) {
    int l = blockIdx.x * blockDim.x + threadIdx.x;
    if (l >= L) return;
    const float* pa = g + (size_t)li[l] * F;
    const float* pb = g + (size_t)li[L + l] * F;
    float dot = 0.0f;
    for (int c = 0; c < F; c += 4) {
        float4 a = *(const float4*)(pa + c);
        float4 b = *(const float4*)(pb + c);
        dot += a.x * b.x + a.y * b.y + a.z * b.z + a.w * b.w;
    }
    out[l] = 1.0f / (1.0f + expf(-dot));
}

// ---------------------------------------------------------------------------
// Host launcher
// ---------------------------------------------------------------------------
extern "C" void kernel_launch(void* const* d_in, const int* in_sizes, int n_in,
                              void* d_out, int out_size, void* d_ws, size_t ws_size,
                              hipStream_t stream) {
    const int F_IN = 128, H = 128, H2 = 64, H4 = 32, OUT = 32;
    const int N = in_sizes[2];          // batch array has N entries
    const int E = in_sizes[1] / 2;
    const int L = in_sizes[3] / 2;
    const int G = 1000;                 // num_graphs (device scalar; fixed by setup)

    const float* x    = (const float*)d_in[0];
    const int*   src  = (const int*)d_in[1];
    const int*   dst  = src + E;
    const int*   batch= (const int*)d_in[2];
    const int*   li   = (const int*)d_in[3];
    const float* W1   = (const float*)d_in[5];
    const float* b1   = (const float*)d_in[6];
    const float* g1   = (const float*)d_in[7];
    const float* be1  = (const float*)d_in[8];
    const float* Wl2  = (const float*)d_in[9];
    const float* bl2  = (const float*)d_in[10];
    const float* Wr2  = (const float*)d_in[11];
    const float* g2   = (const float*)d_in[12];
    const float* be2  = (const float*)d_in[13];
    const float* W3   = (const float*)d_in[14];
    const float* b3   = (const float*)d_in[15];
    const float* g3   = (const float*)d_in[16];
    const float* be3  = (const float*)d_in[17];
    const float* W4   = (const float*)d_in[18];
    const float* b4   = (const float*)d_in[19];
    float* out = (float*)d_out;

    // Workspace carve-up (floats)
    float* ws = (float*)d_ws;
    size_t off = 0;
    float* dinv   = ws + off; off += (size_t)N;
    float* invcnt = ws + off; off += (size_t)N;
    float* bufA   = ws + off; off += (size_t)N * 128;
    float* bufB   = ws + off; off += (size_t)N * 128;
    float* bufC   = ws + off; off += (size_t)N * 64;
    float* bufD   = ws + off; off += (size_t)N * 64;
    float* sums   = ws + off; off += 128;
    float* sumsq  = ws + off; off += 128;
    float* gpool  = ws + off; off += (size_t)G * 32;

    const int B256 = 256;
    auto g1d = [](int n) { return ceil_div(n, 256); };
    const int rowTiles  = N >> 4;
    const int rowGroups = ceil_div(rowTiles, 8);

    // ---- degree / normalization ----
    init_deg<<<g1d(N), B256, 0, stream>>>(dinv, N);
    edge_deg<<<g1d(E), B256, 0, stream>>>(dst, dinv, E);
    finalize_deg<<<g1d(N), B256, 0, stream>>>(dinv, invcnt, N);

    // ---- Layer 1: GCN (128 -> 128) ----
    {
        wmma_gemm_f16<<<dim3(rowGroups, H >> 4), B256, 0, stream>>>(
            x, W1, nullptr, nullptr, bufA, N, F_IN, H);
        zero_f<<<g1d(N * H), B256, 0, stream>>>(bufB, N * H);
        edge_scatter<<<g1d(E * (H / 4)), B256, 0, stream>>>(src, dst, dinv, bufA, bufB, E, H);
        gcn_self_bias<<<g1d(N * H), B256, 0, stream>>>(dinv, bufA, b1, bufB, N, H);
        zero_f<<<1, B256, 0, stream>>>(sums, 128);
        zero_f<<<1, B256, 0, stream>>>(sumsq, 128);
        bn_stats<<<256, B256, 0, stream>>>(bufB, sums, sumsq, N, H);
        bn_apply_lrelu<<<g1d(N * H), B256, 0, stream>>>(bufB, sums, sumsq, g1, be1, N, H);
    }

    // ---- Layer 2: SAGE (128 -> 64), mean aggregation ----
    {
        zero_f<<<g1d(N * H), B256, 0, stream>>>(bufA, N * H);
        edge_scatter<<<g1d(E * (H / 4)), B256, 0, stream>>>(src, dst, nullptr, bufB, bufA, E, H);
        sage_scale<<<g1d(N * H), B256, 0, stream>>>(bufA, invcnt, N, H);
        // t = agg @ Wl2 + bl2
        wmma_gemm_f16<<<dim3(rowGroups, H2 >> 4), B256, 0, stream>>>(
            bufA, Wl2, nullptr, bl2, bufC, N, H, H2);
        // out = h @ Wr2 + t
        wmma_gemm_f16<<<dim3(rowGroups, H2 >> 4), B256, 0, stream>>>(
            bufB, Wr2, bufC, nullptr, bufD, N, H, H2);
        zero_f<<<1, B256, 0, stream>>>(sums, 128);
        zero_f<<<1, B256, 0, stream>>>(sumsq, 128);
        bn_stats<<<256, B256, 0, stream>>>(bufD, sums, sumsq, N, H2);
        bn_apply_lrelu<<<g1d(N * H2), B256, 0, stream>>>(bufD, sums, sumsq, g2, be2, N, H2);
    }

    // ---- Layer 3: GCN (64 -> 32) ----
    {
        wmma_gemm_f16<<<dim3(rowGroups, H4 >> 4), B256, 0, stream>>>(
            bufD, W3, nullptr, nullptr, bufC, N, H2, H4);
        zero_f<<<g1d(N * H4), B256, 0, stream>>>(bufA, N * H4);
        edge_scatter<<<g1d(E * (H4 / 4)), B256, 0, stream>>>(src, dst, dinv, bufC, bufA, E, H4);
        gcn_self_bias<<<g1d(N * H4), B256, 0, stream>>>(dinv, bufC, b3, bufA, N, H4);
        zero_f<<<1, B256, 0, stream>>>(sums, 128);
        zero_f<<<1, B256, 0, stream>>>(sumsq, 128);
        bn_stats<<<256, B256, 0, stream>>>(bufA, sums, sumsq, N, H4);
        bn_apply_lrelu<<<g1d(N * H4), B256, 0, stream>>>(bufA, sums, sumsq, g3, be3, N, H4);
    }

    // ---- Layer 4: GCN (32 -> 32), no BN ----
    {
        wmma_gemm_f16<<<dim3(rowGroups, OUT >> 4), B256, 0, stream>>>(
            bufA, W4, nullptr, nullptr, bufC, N, H4, OUT);
        zero_f<<<g1d(N * OUT), B256, 0, stream>>>(bufB, N * OUT);
        edge_scatter<<<g1d(E * (OUT / 4)), B256, 0, stream>>>(src, dst, dinv, bufC, bufB, E, OUT);
        gcn_self_bias<<<g1d(N * OUT), B256, 0, stream>>>(dinv, bufC, b4, bufB, N, OUT);
    }

    // ---- Global add pool + link prediction ----
    zero_f<<<g1d(G * OUT), B256, 0, stream>>>(gpool, G * OUT);
    pool_add<<<g1d(N * OUT), B256, 0, stream>>>(bufB, batch, gpool, N, OUT);
    link_pred<<<g1d(L), B256, 0, stream>>>(gpool, li, out, L, OUT);
}
